// NeuroUNET_188978561037
// MI455X (gfx1250) — matrix operand order repaired and checked by reading, not compiled, not run on hardware
//
#include <hip/hip_runtime.h>

typedef _Float16 h16;
typedef __attribute__((ext_vector_type(16))) _Float16 v16h;
typedef __attribute__((ext_vector_type(8)))  float    v8f;
typedef __attribute__((ext_vector_type(4)))  unsigned u32x4;

union F16x16 { v16h v; u32x4 q[2]; };

// LDS row stride: 32 payload halfs padded to 40 (80B = 5 x 16B chunks; 5 is
// coprime with the 16 bank-groups so strided b128 accesses spread banks).
#define ROWS 40

// ---------------------------------------------------------------------------
// fp32 -> f16 convert (grid-stride)
// ---------------------------------------------------------------------------
__global__ void k_cvt(const float* __restrict__ s, h16* __restrict__ d, int n) {
    for (int i = blockIdx.x * blockDim.x + threadIdx.x; i < n; i += gridDim.x * blockDim.x)
        d[i] = (h16)s[i];
}

// ---------------------------------------------------------------------------
// conv3x3 (pad=1) implicit GEMM with WMMA f32<-f16. NCHW, B=8.
// cfg 0: 8 waves = 2 cout x 4 w  -> tile 32 cout x 64 px  (W >= 64)
// cfg 1: 8 waves = 4 cout x 2 w  -> tile 64 cout x 32 px  (W == 32)
// Writes pre-BN fp32 y (bias added).
// Fragment layouts (ISA 7.12.2, wave32):
//   A 16x32 f16: lane m=L&15, h=L>>4; VGPR p holds K = (p<4?2p:16+2(p-4))+8h, +1
//     -> per lane: cin [8h,8h+8) and [16+8h,16+8h+8) : two 16B ds_load_b128
//   B 32x16 f16: lane n=L&15, h=L>>4; VGPR p holds K = (p<4?2p:16+2(p-4))+8h, +1
//     -> per lane: cin [8h,8h+8) and [16+8h,16+8h+8) : two 16B ds_load_b128
// ---------------------------------------------------------------------------
__global__ __launch_bounds__(256) void k_conv3x3(
    const h16* __restrict__ in, int Cin, int H, int W,
    const h16* __restrict__ wgt, const float* __restrict__ bias,
    float* __restrict__ y, int Cout, int cfg)
{
    __shared__ __align__(16) h16 s_x[3 * 66 * ROWS];
    __shared__ __align__(16) h16 s_w[9 * 64 * ROWS];
    const int tid  = threadIdx.x;
    const int lane = tid & 31;
    const int wv   = tid >> 5;
    const int WT   = cfg ? 32 : 64;
    const int CT   = cfg ? 64 : 32;
    const int WW   = cfg ? 2 : 4;
    const int ncol = WT + 2;
    const int wq   = wv % WW;
    const int cq   = wv / WW;
    const int w0   = blockIdx.x * WT;
    const int b    = blockIdx.y / H;
    const int h    = blockIdx.y % H;
    const int cout0 = blockIdx.z * CT;
    const int m  = lane & 15;
    const int hh = lane >> 4;
    const int coutL = cq * 16 + m;

    v8f acc;
#pragma unroll
    for (int i = 0; i < 8; i++) acc[i] = 0.0f;

    const int nx = 3 * ncol * 32;
    const int nw = 9 * CT * 32;

    for (int c0 = 0; c0 < Cin; c0 += 32) {
        // stage input patch: [row r][col][cin32] rows padded to ROWS halfs
        for (int e = tid; e < nx; e += 256) {
            int col = e % ncol;
            int t2  = e / ncol;
            int ci  = t2 & 31;
            int r   = t2 >> 5;
            int gc = c0 + ci, gh = h - 1 + r, gw = w0 - 1 + col;
            h16 v = (h16)0.0f;
            if (gc < Cin && (unsigned)gh < (unsigned)H && (unsigned)gw < (unsigned)W)
                v = in[((b * Cin + gc) * H + gh) * W + gw];
            s_x[(r * ncol + col) * ROWS + ci] = v;
        }
        // stage weights: [tap][cout][cin32] rows padded to ROWS halfs
        for (int e = tid; e < nw; e += 256) {
            int t  = e % 9;
            int ci = (e / 9) & 31;
            int co = e / (9 * 32);
            int gc = c0 + ci;
            h16 v = (h16)0.0f;
            if (gc < Cin)
                v = wgt[((cout0 + co) * Cin + gc) * 9 + t];
            s_w[(t * CT + co) * ROWS + ci] = v;
        }
        __syncthreads();
#pragma unroll
        for (int t = 0; t < 9; t++) {
            const int ky = t / 3, kx = t % 3;
            F16x16 af, bf;
            const u32x4* wb = (const u32x4*)(s_w + (t * CT + coutL) * ROWS + 8 * hh);
            af.q[0] = wb[0];   // cin 8h .. 8h+7
            af.q[1] = wb[2];   // cin 16+8h .. 16+8h+7
            const int col = wq * 16 + m + kx;
            const u32x4* xb = (const u32x4*)(s_x + (ky * ncol + col) * ROWS + 8 * hh);
            bf.q[0] = xb[0];   // cin 8h .. 8h+7
            bf.q[1] = xb[2];   // cin 16+8h .. 16+8h+7
            acc = __builtin_amdgcn_wmma_f32_16x16x32_f16(
                false, af.v, false, bf.v, (short)0, acc, false, false);
        }
        __syncthreads();
    }
    const int gw = w0 + wq * 16 + m;
#pragma unroll
    for (int r = 0; r < 8; r++) {
        int co = cout0 + cq * 16 + r + 8 * hh;   // C/D layout: M = r + 8*hh
        y[((b * Cout + co) * H + h) * W + gw] = acc[r] + bias[co];
    }
}

// ---------------------------------------------------------------------------
// conv1x1 GEMM with WMMA: out f16 = W(Cout,Cin) * in(B,Cin,S) + bias
// ---------------------------------------------------------------------------
__global__ __launch_bounds__(256) void k_conv1x1(
    const h16* __restrict__ in, int Cin, int S,
    const h16* __restrict__ wgt, const float* __restrict__ bias,
    h16* __restrict__ out, int Cout)
{
    __shared__ __align__(16) h16 s_x[64 * ROWS];
    __shared__ __align__(16) h16 s_w[32 * ROWS];
    const int tid = threadIdx.x, lane = tid & 31, wv = tid >> 5;
    const int wq = wv & 3, cq = wv >> 2;
    const int s0 = blockIdx.x * 64;
    const int b  = blockIdx.y;
    const int cout0 = blockIdx.z * 32;
    const int m = lane & 15, hh = lane >> 4;
    const int coutL = cq * 16 + m;
    v8f acc;
#pragma unroll
    for (int i = 0; i < 8; i++) acc[i] = 0.0f;

    for (int c0 = 0; c0 < Cin; c0 += 32) {
        for (int e = tid; e < 64 * 32; e += 256) {
            int col = e & 63, ci = e >> 6;
            s_x[col * ROWS + ci] = in[(b * Cin + c0 + ci) * S + s0 + col];
        }
        for (int e = tid; e < 32 * 32; e += 256) {
            int ci = e & 31, co = e >> 5;
            s_w[co * ROWS + ci] = wgt[(cout0 + co) * Cin + c0 + ci];
        }
        __syncthreads();
        F16x16 af, bf;
        const u32x4* wb = (const u32x4*)(s_w + coutL * ROWS + 8 * hh);
        af.q[0] = wb[0];
        af.q[1] = wb[2];
        const u32x4* xb = (const u32x4*)(s_x + (wq * 16 + m) * ROWS + 8 * hh);
        bf.q[0] = xb[0];
        bf.q[1] = xb[2];
        acc = __builtin_amdgcn_wmma_f32_16x16x32_f16(
            false, af.v, false, bf.v, (short)0, acc, false, false);
        __syncthreads();
    }
#pragma unroll
    for (int r = 0; r < 8; r++) {
        int co = cout0 + cq * 16 + r + 8 * hh;
        out[(b * Cout + co) * S + s0 + wq * 16 + m] = (h16)(acc[r] + bias[co]);
    }
}

// ---------------------------------------------------------------------------
// per-channel sum/sumsq over pre-BN fp32 y (B=8)
// ---------------------------------------------------------------------------
__global__ void k_stats(const float* __restrict__ y, int C, int HW,
                        float* __restrict__ ssum, float* __restrict__ ssq)
{
    const int c = blockIdx.y;
    const int N = 8 * HW;
    int base = blockIdx.x * 8192 + threadIdx.x;
    float s1 = 0.f, s2 = 0.f;
    for (int k = 0; k < 32; k++) {
        int n = base + k * 256;
        if (n < N) {
            int b = n / HW, s = n - b * HW;
            float v = y[(b * C + c) * HW + s];
            s1 += v; s2 += v * v;
        }
    }
    for (int off = 1; off < 32; off <<= 1) {
        s1 += __shfl_xor(s1, off, 32);
        s2 += __shfl_xor(s2, off, 32);
    }
    if ((threadIdx.x & 31) == 0) {
        atomicAdd(&ssum[c], s1);
        atomicAdd(&ssq[c], s2);
    }
}

__global__ void k_bnfin(const float* __restrict__ ssum, const float* __restrict__ ssq,
                        const float* __restrict__ g, const float* __restrict__ bt,
                        int C, float invN, float* __restrict__ scale, float* __restrict__ shift)
{
    int c = blockIdx.x * blockDim.x + threadIdx.x;
    if (c >= C) return;
    float mean = ssum[c] * invN;
    float var  = ssq[c] * invN - mean * mean;
    float sc   = g[c] * rsqrtf(var + 1e-5f);
    scale[c] = sc;
    shift[c] = bt[c] - mean * sc;
}

__global__ void k_bnapply(const float* __restrict__ y, const float* __restrict__ scale,
                          const float* __restrict__ shift, h16* __restrict__ act,
                          int n, int HW, int C)
{
    for (int i = blockIdx.x * blockDim.x + threadIdx.x; i < n; i += gridDim.x * blockDim.x) {
        int c = (i / HW) % C;
        float v = y[i] * scale[c] + shift[c];
        act[i] = (h16)fmaxf(v, 0.0f);
    }
}

// ---------------------------------------------------------------------------
// Haar DWT: (8,C,H,W) -> (8,4C,H/2,W/2), channel 4c+{A,H,V,D}
// ---------------------------------------------------------------------------
__global__ void k_dwt(const h16* __restrict__ in, h16* __restrict__ out,
                      int C, int H, int W)
{
    const int H2 = H / 2, W2 = W / 2;
    const int n = 8 * C * H2 * W2;
    for (int i = blockIdx.x * blockDim.x + threadIdx.x; i < n; i += gridDim.x * blockDim.x) {
        int j = i % W2; int t = i / W2;
        int ii = t % H2; t /= H2;
        int c = t % C; int b = t / C;
        const h16* p = in + ((b * C + c) * H + 2 * ii) * W + 2 * j;
        float a  = (float)p[0], bb = (float)p[1];
        float cc = (float)p[W], dd = (float)p[W + 1];
        int base = ((b * 4 * C + 4 * c) * H2 + ii) * W2 + j;
        int step = H2 * W2;
        out[base]            = (h16)((a + bb + cc + dd) * 0.5f);
        out[base + step]     = (h16)((a + bb - cc - dd) * 0.5f);
        out[base + 2 * step] = (h16)((a - bb + cc - dd) * 0.5f);
        out[base + 3 * step] = (h16)((a - bb - cc + dd) * 0.5f);
    }
}

// Haar IWT: (8,4C,H,W) -> channels [0,C) of (8,Ctot,2H,2W)
__global__ void k_iwt(const h16* __restrict__ in, h16* __restrict__ out,
                      int C, int H, int W, int Ctot)
{
    const int n = 8 * C * H * W;
    for (int i = blockIdx.x * blockDim.x + threadIdx.x; i < n; i += gridDim.x * blockDim.x) {
        int j = i % W; int t = i / W;
        int ii = t % H; t /= H;
        int c = t % C; int b = t / C;
        int base = ((b * 4 * C + 4 * c) * H + ii) * W + j;
        int step = H * W;
        float cA = (float)in[base];
        float cH = (float)in[base + step];
        float cV = (float)in[base + 2 * step];
        float cD = (float)in[base + 3 * step];
        float a  = (cA + cH + cV + cD) * 0.5f;
        float bb = (cA + cH - cV - cD) * 0.5f;
        float cc = (cA - cH + cV - cD) * 0.5f;
        float dd = (cA - cH - cV + cD) * 0.5f;
        h16* o = out + ((b * Ctot + c) * (2 * H) + 2 * ii) * (2 * W) + 2 * j;
        o[0] = (h16)a; o[1] = (h16)bb;
        o[2 * W] = (h16)cc; o[2 * W + 1] = (h16)dd;
    }
}

// copy skip activations (8,C,HW) into concat buffer at channel offset
__global__ void k_copych(const h16* __restrict__ in, h16* __restrict__ out,
                         int C, int HW, int Coff, int Ctot)
{
    const int n = 8 * C * HW;
    for (int i = blockIdx.x * blockDim.x + threadIdx.x; i < n; i += gridDim.x * blockDim.x) {
        int s = i % HW; int t = i / HW;
        int c = t % C; int b = t / C;
        out[(b * Ctot + Coff + c) * HW + s] = in[i];
    }
}

// final 1x1 conv 64 -> 2, fp32 output
__global__ void k_outconv(const h16* __restrict__ in, const float* __restrict__ w,
                          const float* __restrict__ bias, float* __restrict__ out, int S)
{
    int i = blockIdx.x * blockDim.x + threadIdx.x;
    if (i >= 8 * S) return;
    int b = i / S, s = i - b * S;
    const h16* ip = in + (b * 64) * S + s;
    float a0 = bias[0], a1 = bias[1];
    for (int c = 0; c < 64; c++) {
        float v = (float)ip[c * S];
        a0 += w[c] * v;
        a1 += w[64 + c] * v;
    }
    out[(b * 2 + 0) * S + s] = a0;
    out[(b * 2 + 1) * S + s] = a1;
}

// ---------------------------------------------------------------------------
// host orchestration
// ---------------------------------------------------------------------------
enum {
    CX = 0,
    E1_W1, E1_B1, E1_G1, E1_BT1, E1_W2, E1_B2, E1_G2, E1_BT2,
    A1_W, A1_B,
    E2_W1, E2_B1, E2_G1, E2_BT1, E2_W2, E2_B2, E2_G2, E2_BT2,
    A2_W, A2_B,
    E3_W1, E3_B1, E3_G1, E3_BT1, E3_W2, E3_B2, E3_G2, E3_BT2,
    A3_W, A3_B,
    BO_W1, BO_B1, BO_G1, BO_BT1, BO_W2, BO_B2, BO_G2, BO_BT2,
    X3_W, X3_B,
    D3_W1, D3_B1, D3_G1, D3_BT1, D3_W2, D3_B2, D3_G2, D3_BT2,
    X2_W, X2_B,
    D2_W1, D2_B1, D2_G1, D2_BT1, D2_W2, D2_B2, D2_G2, D2_BT2,
    X1_W, X1_B,
    D1_W1, D1_B1, D1_G1, D1_BT1, D1_W2, D1_B2, D1_G2, D1_BT2,
    OUT_W, OUT_B, NPARAM
};

// canonical (insertion-order) index -> position under jax-sorted flattening
static const int mapB[NPARAM] = {
    70,
    44, 38, 42, 40, 45, 39, 43, 41,
    1, 0,
    52, 46, 50, 48, 53, 47, 51, 49,
    3, 2,
    60, 54, 58, 56, 61, 55, 59, 57,
    5, 4,
    12, 6, 10, 8, 13, 7, 11, 9,
    67, 66,
    36, 30, 34, 32, 37, 31, 35, 33,
    65, 64,
    28, 22, 26, 24, 29, 23, 27, 25,
    63, 62,
    20, 14, 18, 16, 21, 15, 19, 17,
    69, 68
};

extern "C" void kernel_launch(void* const* d_in, const int* in_sizes, int n_in,
                              void* d_out, int out_size, void* d_ws, size_t ws_size,
                              hipStream_t stream)
{
    if (n_in < NPARAM) return;
    const float* P[NPARAM];
    const bool orderA = (in_sizes[0] == 8 * 2 * 256 * 256);
    for (int c = 0; c < NPARAM; c++) P[c] = (const float*)d_in[orderA ? c : mapB[c]];

    const size_t MiB = 1ull << 20;
    char* ws = (char*)d_ws;
    h16*   W16base = (h16*)ws;                       // [0, 32 MiB): f16 weights
    float* ssum    = (float*)(ws + 32 * MiB);        // stats
    float* ssq     = ssum + 4096;
    float* sscale  = ssq + 4096;
    float* sshift  = sscale + 4096;
    h16*   X16  = (h16*)(ws + 34 * MiB);             // 2 MiB
    h16*   ENC1 = (h16*)(ws + 36 * MiB);             // 64 MiB
    h16*   ENC2 = (h16*)(ws + 100 * MiB);            // 32 MiB
    h16*   ENC3 = (h16*)(ws + 132 * MiB);            // 16 MiB
    float* Yb   = (float*)(ws + 160 * MiB);          // 128 MiB
    h16*   T1   = (h16*)(ws + 288 * MiB);            // 128 MiB
    h16*   T2   = (h16*)(ws + 416 * MiB);            // 128 MiB
    h16*   T3   = (h16*)(ws + 544 * MiB);            // 128 MiB (concat)

    size_t woff = 0;
    auto cvtw = [&](int canon, int elems) -> const h16* {
        h16* dst = W16base + woff;
        woff += (size_t)((elems + 127) & ~127);
        int blocks = (elems + 1023) / 1024; if (blocks < 1) blocks = 1;
        k_cvt<<<blocks, 256, 0, stream>>>(P[canon], dst, elems);
        return dst;
    };

    const h16* wE1a = cvtw(E1_W1, 64 * 2 * 9);
    const h16* wE1b = cvtw(E1_W2, 64 * 64 * 9);
    const h16* wA1  = cvtw(A1_W, 64 * 256);
    const h16* wE2a = cvtw(E2_W1, 128 * 64 * 9);
    const h16* wE2b = cvtw(E2_W2, 128 * 128 * 9);
    const h16* wA2  = cvtw(A2_W, 128 * 512);
    const h16* wE3a = cvtw(E3_W1, 256 * 128 * 9);
    const h16* wE3b = cvtw(E3_W2, 256 * 256 * 9);
    const h16* wA3  = cvtw(A3_W, 256 * 1024);
    const h16* wBOa = cvtw(BO_W1, 512 * 256 * 9);
    const h16* wBOb = cvtw(BO_W2, 512 * 512 * 9);
    const h16* wX3  = cvtw(X3_W, 1024 * 512);
    const h16* wD3a = cvtw(D3_W1, 256 * 512 * 9);
    const h16* wD3b = cvtw(D3_W2, 256 * 256 * 9);
    const h16* wX2  = cvtw(X2_W, 512 * 256);
    const h16* wD2a = cvtw(D2_W1, 128 * 256 * 9);
    const h16* wD2b = cvtw(D2_W2, 128 * 128 * 9);
    const h16* wX1  = cvtw(X1_W, 256 * 128);
    const h16* wD1a = cvtw(D1_W1, 64 * 128 * 9);
    const h16* wD1b = cvtw(D1_W2, 64 * 64 * 9);

    k_cvt<<<1024, 256, 0, stream>>>(P[CX], X16, 8 * 2 * 256 * 256);

    auto conv3 = [&](const h16* in, int Cin, int H, int W,
                     const h16* wgt, int bias_c, int Cout) {
        int cfg = (W >= 64) ? 0 : 1;
        int WT = cfg ? 32 : 64, CT = cfg ? 64 : 32;
        dim3 grid(W / WT, 8 * H, Cout / CT);
        k_conv3x3<<<grid, 256, 0, stream>>>(in, Cin, H, W, wgt, P[bias_c], Yb, Cout, cfg);
    };
    auto bnrelu = [&](int Cout, int H, int W, int g_c, int bt_c, h16* act) {
        hipMemsetAsync(ssum, 0, Cout * sizeof(float), stream);
        hipMemsetAsync(ssq, 0, Cout * sizeof(float), stream);
        int HW = H * W, N = 8 * HW;
        dim3 sgrid((N + 8191) / 8192, Cout);
        k_stats<<<sgrid, 256, 0, stream>>>(Yb, Cout, HW, ssum, ssq);
        k_bnfin<<<(Cout + 255) / 256, 256, 0, stream>>>(ssum, ssq, P[g_c], P[bt_c],
                                                        Cout, 1.0f / (float)N, sscale, sshift);
        int n = 8 * Cout * HW;
        k_bnapply<<<4096, 256, 0, stream>>>(Yb, sscale, sshift, act, n, HW, Cout);
    };
    auto conv1 = [&](const h16* in, int Cin, int S, const h16* wgt, int bias_c,
                     h16* out, int Cout) {
        dim3 grid(S / 64, 8, Cout / 32);
        k_conv1x1<<<grid, 256, 0, stream>>>(in, Cin, S, wgt, P[bias_c], out, Cout);
    };

    // ---------------- encoder ----------------
    conv3(X16, 2, 256, 256, wE1a, E1_B1, 64);   bnrelu(64, 256, 256, E1_G1, E1_BT1, T1);
    conv3(T1, 64, 256, 256, wE1b, E1_B2, 64);   bnrelu(64, 256, 256, E1_G2, E1_BT2, ENC1);
    k_dwt<<<4096, 256, 0, stream>>>(ENC1, T1, 64, 256, 256);        // (8,256,128,128)
    conv1(T1, 256, 128 * 128, wA1, A1_B, T2, 64);                   // (8,64,128,128)

    conv3(T2, 64, 128, 128, wE2a, E2_B1, 128);  bnrelu(128, 128, 128, E2_G1, E2_BT1, T1);
    conv3(T1, 128, 128, 128, wE2b, E2_B2, 128); bnrelu(128, 128, 128, E2_G2, E2_BT2, ENC2);
    k_dwt<<<4096, 256, 0, stream>>>(ENC2, T1, 128, 128, 128);       // (8,512,64,64)
    conv1(T1, 512, 64 * 64, wA2, A2_B, T2, 128);                    // (8,128,64,64)

    conv3(T2, 128, 64, 64, wE3a, E3_B1, 256);   bnrelu(256, 64, 64, E3_G1, E3_BT1, T1);
    conv3(T1, 256, 64, 64, wE3b, E3_B2, 256);   bnrelu(256, 64, 64, E3_G2, E3_BT2, ENC3);
    k_dwt<<<4096, 256, 0, stream>>>(ENC3, T1, 256, 64, 64);         // (8,1024,32,32)
    conv1(T1, 1024, 32 * 32, wA3, A3_B, T2, 256);                   // (8,256,32,32)

    // ---------------- bottleneck ----------------
    conv3(T2, 256, 32, 32, wBOa, BO_B1, 512);   bnrelu(512, 32, 32, BO_G1, BO_BT1, T1);
    conv3(T1, 512, 32, 32, wBOb, BO_B2, 512);   bnrelu(512, 32, 32, BO_G2, BO_BT2, T2);

    // ---------------- decoder ----------------
    conv1(T2, 512, 32 * 32, wX3, X3_B, T1, 1024);                   // (8,1024,32,32)
    k_iwt<<<4096, 256, 0, stream>>>(T1, T3, 256, 32, 32, 512);      // concat3 lo
    k_copych<<<4096, 256, 0, stream>>>(ENC3, T3, 256, 64 * 64, 256, 512);
    conv3(T3, 512, 64, 64, wD3a, D3_B1, 256);   bnrelu(256, 64, 64, D3_G1, D3_BT1, T1);
    conv3(T1, 256, 64, 64, wD3b, D3_B2, 256);   bnrelu(256, 64, 64, D3_G2, D3_BT2, T2);

    conv1(T2, 256, 64 * 64, wX2, X2_B, T1, 512);                    // (8,512,64,64)
    k_iwt<<<4096, 256, 0, stream>>>(T1, T3, 128, 64, 64, 256);      // concat2 lo
    k_copych<<<4096, 256, 0, stream>>>(ENC2, T3, 128, 128 * 128, 128, 256);
    conv3(T3, 256, 128, 128, wD2a, D2_B1, 128); bnrelu(128, 128, 128, D2_G1, D2_BT1, T1);
    conv3(T1, 128, 128, 128, wD2b, D2_B2, 128); bnrelu(128, 128, 128, D2_G2, D2_BT2, T2);

    conv1(T2, 128, 128 * 128, wX1, X1_B, T1, 256);                  // (8,256,128,128)
    k_iwt<<<4096, 256, 0, stream>>>(T1, T3, 64, 128, 128, 128);     // concat1 lo
    k_copych<<<4096, 256, 0, stream>>>(ENC1, T3, 64, 256 * 256, 64, 128);
    conv3(T3, 128, 256, 256, wD1a, D1_B1, 64);  bnrelu(64, 256, 256, D1_G1, D1_BT1, T1);
    conv3(T1, 64, 256, 256, wD1b, D1_B2, 64);   bnrelu(64, 256, 256, D1_G2, D1_BT2, T2);

    // ---------------- head ----------------
    k_outconv<<<(8 * 256 * 256 + 255) / 256, 256, 0, stream>>>(
        T2, P[OUT_W], P[OUT_B], (float*)d_out, 256 * 256);

    (void)out_size; (void)ws_size;
}
